// GCNEncoder_21543555956944
// MI455X (gfx1250) — compile-verified, matrix operand order checked by reading, not run on hardware
//
#include <hip/hip_runtime.h>
#include <math.h>

// ---------------------------------------------------------------------------
// 2-layer single-head GATConv (PyG semantics, add_self_loops=False)
//   h   = x @ W                     (bf16 WMMA, f32 accumulate, W pre-transposed)
//   a_s = h @ a_src ; a_d = h @ a_dst
//   e   = leaky_relu(a_s[src] + a_d[dst], 0.2)
//   softmax over edges grouped by dst (segment max / exp / segment sum)
//   out[dst] += alpha * h[src]  ;  out += bias ; (relu after layer 1)
// ---------------------------------------------------------------------------

typedef __attribute__((ext_vector_type(16))) __bf16 v16bf;
typedef __attribute__((ext_vector_type(8)))  __bf16 v8bf;
typedef __attribute__((ext_vector_type(8)))  float  v8f;

#define NEG_SLOPE 0.2f

__device__ __forceinline__ unsigned short f32_to_bf16_rne(float f) {
  unsigned int u = __float_as_uint(f);
  u += 0x7FFFu + ((u >> 16) & 1u);       // round-to-nearest-even
  return (unsigned short)(u >> 16);
}

// exact float atomic max via sign-aware integer atomics
__device__ __forceinline__ void atomicMaxF32(float* addr, float val) {
  if (val >= 0.0f) atomicMax((int*)addr, __float_as_int(val));
  else             atomicMin((unsigned int*)addr, __float_as_uint(val));
}

// ---------------------------------------------------------------------------

__global__ void fill_f32x4_kernel(float4* __restrict__ p, float v, int n4) {
  int i = blockIdx.x * blockDim.x + threadIdx.x;
  if (i < n4) p[i] = make_float4(v, v, v, v);
}

// 4 elements / thread (n divisible by 4 for all our tensors)
__global__ void cvt_f32_to_bf16_kernel(const float4* __restrict__ in,
                                       ushort4* __restrict__ out, int n4) {
  int i = blockIdx.x * blockDim.x + threadIdx.x;
  if (i >= n4) return;
  float4 v = in[i];
  ushort4 o;
  o.x = f32_to_bf16_rne(v.x); o.y = f32_to_bf16_rne(v.y);
  o.z = f32_to_bf16_rne(v.z); o.w = f32_to_bf16_rne(v.w);
  out[i] = o;
}

// W[K,N] f32 row-major -> Wt[N,K] bf16 (column-major W). Tiny matrices.
__global__ void cvt_transpose_bf16_kernel(const float* __restrict__ W,
                                          unsigned short* __restrict__ Wt,
                                          int K, int N) {
  int i = blockIdx.x * blockDim.x + threadIdx.x;
  if (i >= K * N) return;
  int k = i / N, n = i - k * N;
  Wt[(size_t)n * K + k] = f32_to_bf16_rne(W[i]);
}

// C[M,N] = A[M,K] * B[K,N] with Bt = B^T stored [N,K] bf16 row-major.
// blockDim = 256 (8 waves); one 16x16 output tile per wave; K templated -> full unroll.
template <int K>
__global__ __launch_bounds__(256) void gemm_wmma_bf16_nt_kernel(
    const unsigned short* __restrict__ Au, const unsigned short* __restrict__ Btu,
    float* __restrict__ C, int M, int N)
{
  const __bf16* A  = (const __bf16*)Au;
  const __bf16* Bt = (const __bf16*)Btu;
  const int tid  = threadIdx.x;
  const int lane = tid & 31;
  const int wave = tid >> 5;
  const int ntiles = N >> 4;
  const long long tiles = (long long)(M >> 4) * ntiles;
  const long long w = (long long)blockIdx.x * 8 + wave;
  if (w >= tiles) return;                 // wave-uniform exit, EXEC stays all-ones
  const int mt   = (int)(w / ntiles);
  const int nt   = (int)(w % ntiles);
  const int half = lane >> 4;             // K-half selector for fragments
  const int l16  = lane & 15;             // row (A) / column (B) / column (D)

  // A: lane l16 holds row mt*16+l16 ; Bt: lane l16 holds column nt*16+l16.
  // Per k-step of 32: lanes 0-15 carry K[0..7]+[16..23], lanes 16-31 K[8..15]+[24..31].
  const __bf16* arow = A  + (size_t)(mt * 16 + l16) * K;
  const __bf16* brow = Bt + (size_t)(nt * 16 + l16) * K;

  v8f acc = {};
#pragma unroll
  for (int kt = 0; kt < K; kt += 32) {
    v8bf a0 = *(const v8bf*)(arow + kt + half * 8);
    v8bf a1 = *(const v8bf*)(arow + kt + 16 + half * 8);
    v8bf b0 = *(const v8bf*)(brow + kt + half * 8);
    v8bf b1 = *(const v8bf*)(brow + kt + 16 + half * 8);
    v16bf a, b;
#pragma unroll
    for (int e = 0; e < 8; ++e) {
      a[e] = a0[e]; a[8 + e] = a1[e];
      b[e] = b0[e]; b[8 + e] = b1[e];
    }
    acc = __builtin_amdgcn_wmma_f32_16x16x32_bf16(
        false, a, false, b, (short)0, acc, false, false);
  }
  // D layout: VGPR r -> row half*8 + r ; lane&15 -> column
  float* crow = C + (size_t)(mt * 16 + half * 8) * N + nt * 16 + l16;
#pragma unroll
  for (int r = 0; r < 8; ++r) crow[(size_t)r * N] = acc[r];
}

// per-node dots: os[n] = H[n,:]·avs ; od[n] = H[n,:]·avd  (one wave per node)
__global__ __launch_bounds__(256) void row_dots_kernel(
    const float* __restrict__ H, const float* __restrict__ avs,
    const float* __restrict__ avd, float* __restrict__ os,
    float* __restrict__ od, int Nn, int D)
{
  int wid  = (int)(((long long)blockIdx.x * blockDim.x + threadIdx.x) >> 5);
  int lane = threadIdx.x & 31;
  if (wid >= Nn) return;
  float ss = 0.f, sd = 0.f;
  for (int c = lane; c < D; c += 32) {
    float hv = H[(size_t)wid * D + c];
    ss += hv * avs[c];
    sd += hv * avd[c];
  }
#pragma unroll
  for (int off = 16; off > 0; off >>= 1) {
    ss += __shfl_xor(ss, off, 32);
    sd += __shfl_xor(sd, off, 32);
  }
  if (lane == 0) { os[wid] = ss; od[wid] = sd; }
}

__global__ void edge_logits_kernel(const int* __restrict__ src, const int* __restrict__ dst,
                                   const float* __restrict__ os, const float* __restrict__ od,
                                   float* __restrict__ ebuf, float* __restrict__ maxb, int E)
{
  int i = blockIdx.x * blockDim.x + threadIdx.x;
  if (i >= E) return;
  float e = os[src[i]] + od[dst[i]];
  e = (e > 0.f) ? e : NEG_SLOPE * e;
  ebuf[i] = e;
  atomicMaxF32(&maxb[dst[i]], e);
}

__global__ void edge_exp_kernel(const int* __restrict__ dst, float* __restrict__ ebuf,
                                const float* __restrict__ maxb, float* __restrict__ den, int E)
{
  int i = blockIdx.x * blockDim.x + threadIdx.x;
  if (i >= E) return;
  int d = dst[i];
  float ex = __expf(ebuf[i] - maxb[d]);
  ebuf[i] = ex;
  atomicAdd(&den[d], ex);
}

// out[dst, c..c+3] += alpha * H[src, c..c+3] ; one thread per (edge, 4 channels).
// logQ = log2(D/4). h[src] row gathered as float4 (L2-resident), scalar f32 atomics out.
__global__ __launch_bounds__(256) void aggregate4_kernel(
    const int* __restrict__ src, const int* __restrict__ dst,
    const float* __restrict__ ex, const float* __restrict__ den,
    const float* __restrict__ H, float* __restrict__ out, int E, int logQ)
{
  long long gid = (long long)blockIdx.x * blockDim.x + threadIdx.x;
  if (gid >= ((long long)E << logQ)) return;
  int e = (int)(gid >> logQ);
  int q = (int)(gid & ((1 << logQ) - 1));
  int s = src[e], d = dst[e];
  float alpha = ex[e] / (den[d] + 1e-16f);
  const int logD = logQ + 2;
  float4 hv = *(const float4*)(H + ((size_t)s << logD) + (q << 2));
  float* o = out + ((size_t)d << logD) + (q << 2);
  atomicAdd(o + 0, alpha * hv.x);
  atomicAdd(o + 1, alpha * hv.y);
  atomicAdd(o + 2, alpha * hv.z);
  atomicAdd(o + 3, alpha * hv.w);
}

// layer-1 epilogue: relu(agg + bias) -> bf16 input for layer-2 GEMM
__global__ void bias_relu_bf16_kernel(const float* __restrict__ in,
                                      const float* __restrict__ bias,
                                      unsigned short* __restrict__ outb, int n, int Dmask)
{
  int i = blockIdx.x * blockDim.x + threadIdx.x;
  if (i >= n) return;
  float v = in[i] + bias[i & Dmask];
  outb[i] = f32_to_bf16_rne(fmaxf(v, 0.f));
}

__global__ void bias_add_kernel(float* __restrict__ out, const float* __restrict__ bias,
                                int n, int Dmask)
{
  int i = blockIdx.x * blockDim.x + threadIdx.x;
  if (i < n) out[i] += bias[i & Dmask];
}

// ---------------------------------------------------------------------------

extern "C" void kernel_launch(void* const* d_in, const int* in_sizes, int n_in,
                              void* d_out, int out_size, void* d_ws, size_t ws_size,
                              hipStream_t stream)
{
  const float* x   = (const float*)d_in[0];
  const int*   ei  = (const int*)d_in[1];
  const float* W1  = (const float*)d_in[2];
  const float* as1 = (const float*)d_in[3];
  const float* ad1 = (const float*)d_in[4];
  const float* b1  = (const float*)d_in[5];
  const float* W2  = (const float*)d_in[6];
  const float* as2 = (const float*)d_in[7];
  const float* ad2 = (const float*)d_in[8];
  const float* b2  = (const float*)d_in[9];
  float* out = (float*)d_out;
  (void)n_in; (void)out_size; (void)ws_size;

  const int DIN = 128, DH = 128, DOUT = 64;
  const int Nn = in_sizes[0] / DIN;      // 100000
  const int E  = in_sizes[1] / 2;        // 1600000
  const int* src = ei;
  const int* dst = ei + E;

  // workspace bump allocator (256B aligned slabs)
  char* p = (char*)d_ws;
  auto bump = [&](size_t bytes) -> char* {
    char* r = p; p += (bytes + 255) & ~(size_t)255; return r;
  };
  unsigned short* Xb1  = (unsigned short*)bump((size_t)Nn * DIN * 2);
  unsigned short* W1bt = (unsigned short*)bump((size_t)DIN * DH * 2);
  unsigned short* W2bt = (unsigned short*)bump((size_t)DH * DOUT * 2);
  float* H1   = (float*)bump((size_t)Nn * DH * 4);
  float* agg1 = (float*)bump((size_t)Nn * DH * 4);
  unsigned short* Xb2 = (unsigned short*)bump((size_t)Nn * DH * 2);
  float* H2   = (float*)bump((size_t)Nn * DOUT * 4);
  float* osrc = (float*)bump((size_t)Nn * 4);
  float* odst = (float*)bump((size_t)Nn * 4);
  float* maxb = (float*)bump((size_t)Nn * 4);
  float* den  = (float*)bump((size_t)Nn * 4);
  float* ebuf = (float*)bump((size_t)E * 4);

  const int TB = 256;
  const float NINF = -__builtin_inff();
#define GRID(n) ((unsigned)(((long long)(n) + TB - 1) / TB))

  // ---- precision conversion ------------------------------------------------
  cvt_f32_to_bf16_kernel<<<GRID(Nn * DIN / 4), TB, 0, stream>>>(
      (const float4*)x, (ushort4*)Xb1, Nn * DIN / 4);
  cvt_transpose_bf16_kernel<<<GRID(DIN * DH), TB, 0, stream>>>(W1, W1bt, DIN, DH);
  cvt_transpose_bf16_kernel<<<GRID(DH * DOUT), TB, 0, stream>>>(W2, W2bt, DH, DOUT);

  // ---- layer 1 -------------------------------------------------------------
  {
    unsigned gblocks = (unsigned)(((long long)(Nn / 16) * (DH / 16) + 7) / 8);
    gemm_wmma_bf16_nt_kernel<128><<<gblocks, TB, 0, stream>>>(Xb1, W1bt, H1, Nn, DH);
    row_dots_kernel<<<GRID((long long)Nn * 32), TB, 0, stream>>>(H1, as1, ad1, osrc, odst, Nn, DH);
    fill_f32x4_kernel<<<GRID(Nn / 4), TB, 0, stream>>>((float4*)maxb, NINF, Nn / 4);
    fill_f32x4_kernel<<<GRID(Nn / 4), TB, 0, stream>>>((float4*)den, 0.f, Nn / 4);
    fill_f32x4_kernel<<<GRID(Nn * DH / 4), TB, 0, stream>>>((float4*)agg1, 0.f, Nn * DH / 4);
    edge_logits_kernel<<<GRID(E), TB, 0, stream>>>(src, dst, osrc, odst, ebuf, maxb, E);
    edge_exp_kernel<<<GRID(E), TB, 0, stream>>>(dst, ebuf, maxb, den, E);
    aggregate4_kernel<<<GRID((long long)E * (DH / 4)), TB, 0, stream>>>(
        src, dst, ebuf, den, H1, agg1, E, 5);   // logQ = log2(128/4)
    bias_relu_bf16_kernel<<<GRID(Nn * DH), TB, 0, stream>>>(agg1, b1, Xb2, Nn * DH, DH - 1);
  }

  // ---- layer 2 -------------------------------------------------------------
  {
    unsigned gblocks = (unsigned)(((long long)(Nn / 16) * (DOUT / 16) + 7) / 8);
    gemm_wmma_bf16_nt_kernel<128><<<gblocks, TB, 0, stream>>>(Xb2, W2bt, H2, Nn, DOUT);
    row_dots_kernel<<<GRID((long long)Nn * 32), TB, 0, stream>>>(H2, as2, ad2, osrc, odst, Nn, DOUT);
    fill_f32x4_kernel<<<GRID(Nn / 4), TB, 0, stream>>>((float4*)maxb, NINF, Nn / 4);
    fill_f32x4_kernel<<<GRID(Nn / 4), TB, 0, stream>>>((float4*)den, 0.f, Nn / 4);
    fill_f32x4_kernel<<<GRID(Nn * DOUT / 4), TB, 0, stream>>>((float4*)out, 0.f, Nn * DOUT / 4);
    edge_logits_kernel<<<GRID(E), TB, 0, stream>>>(src, dst, osrc, odst, ebuf, maxb, E);
    edge_exp_kernel<<<GRID(E), TB, 0, stream>>>(dst, ebuf, maxb, den, E);
    aggregate4_kernel<<<GRID((long long)E * (DOUT / 4)), TB, 0, stream>>>(
        src, dst, ebuf, den, H2, out, E, 4);    // logQ = log2(64/4)
    bias_add_kernel<<<GRID(Nn * DOUT), TB, 0, stream>>>(out, b2, Nn * DOUT, DOUT - 1);
  }
#undef GRID
}
